// ATMOp_34454227649120
// MI455X (gfx1250) — matrix operand order repaired and checked by reading, not compiled, not run on hardware
//
#include <hip/hip_runtime.h>

// Problem dims (compile-time constants from the reference)
constexpr int kB   = 8;
constexpr int kC   = 512;   // in-channels == GEMM K
constexpr int kN   = 4096;  // sequence length == GEMM N
constexpr int kOUT = 512;   // out-channels == GEMM M

typedef __attribute__((ext_vector_type(16))) _Float16 v16h;
typedef __attribute__((ext_vector_type(8)))  float    v8f;

union frag16 { v16h v; uint4 q[2]; };

// Async global->LDS copy path (gfx1250): guard on builtin availability so the
// file compiles on both toolchains; fall back to the register-staging pipeline.
#if defined(__HIP_DEVICE_COMPILE__) && \
    __has_builtin(__builtin_amdgcn_global_load_async_to_lds_b128)
#define USE_ASYNC_COPY 1
#else
#define USE_ASYNC_COPY 0
#endif

#if USE_ASYNC_COPY
// Builtin signature (from the compiler diagnostic): parameters are
// "int4 __device__ *" (AS1) and "int4 __shared__ *" (AS3) where int4 is
// int __attribute__((vector_size(16))).  Low 32 bits of a generic LDS
// pointer are the hardware LDS byte address (ISA §10.2 aperture mapping),
// so inttoptr is equivalent to addrspacecast.
typedef int v4i_t __attribute__((vector_size(16)));
typedef __attribute__((address_space(1))) v4i_t as1_v4i;
typedef __attribute__((address_space(3))) v4i_t as3_v4i;
__device__ __forceinline__ as1_v4i* to_global(const void* p) {
    return (as1_v4i*)(reinterpret_cast<uintptr_t>(p));
}
__device__ __forceinline__ as3_v4i* to_lds(const void* p) {
    return (as3_v4i*)(reinterpret_cast<uintptr_t>(p));
}
#endif

// ---------------------------------------------------------------------------
// Pass 0: weight f32 -> f16 (row-major [OUT][C])
// ---------------------------------------------------------------------------
__global__ void wcvt_kernel(const float* __restrict__ w, _Float16* __restrict__ wh) {
    const int i = blockIdx.x * blockDim.x + threadIdx.x;
    wh[i] = (_Float16)w[i];
}

// ---------------------------------------------------------------------------
// Pass 1: deformable 1-D bilinear sampling + attention mask, f32 -> f16,
// stored TRANSPOSED as St[b][n][c] (K-contiguous for the GEMM B operand).
// 64(c) x 64(n) tile per block, LDS transpose so reads coalesce over n and
// writes coalesce over c.
// ---------------------------------------------------------------------------
__global__ __launch_bounds__(256) void sample_kernel(const float* __restrict__ x,
                                                     const float* __restrict__ off,
                                                     const int*   __restrict__ mask,
                                                     _Float16*    __restrict__ St) {
    __shared__ _Float16 tile[64][65];
    const int b  = blockIdx.z;
    const int c0 = blockIdx.y * 64;
    const int n0 = blockIdx.x * 64;
    const int t  = threadIdx.x;

    const int   nn    = t & 63;
    const int   n     = n0 + nn;
    const float mval  = (float)mask[(size_t)b * kN + n];
    const float nposf = (float)n;

    #pragma unroll
    for (int j = 0; j < 16; ++j) {
        const int    cc   = (t >> 6) + j * 4;
        const size_t base = ((size_t)b * kC + (size_t)(c0 + cc)) * kN;
        const float  p    = nposf + off[base + n];
        const float  fl   = floorf(p);
        const int    li   = (int)fl;
        const float  fr   = p - fl;
        const int    hi   = li + 1;
        const float  v0   = (li >= 0 && li < kN) ? x[base + li] : 0.0f;
        const float  v1   = (hi >= 0 && hi < kN) ? x[base + hi] : 0.0f;
        const float  s    = fmaf(fr, v1 - v0, v0) * mval;   // v0*(1-fr)+v1*fr
        tile[cc][nn] = (_Float16)s;
    }
    __syncthreads();
    const int cc = t & 63;
    #pragma unroll
    for (int j = 0; j < 16; ++j) {
        const int nw = (t >> 6) + j * 4;
        St[((size_t)b * kN + (size_t)(n0 + nw)) * kC + (c0 + cc)] = tile[cc][nw];
    }
}

// ---------------------------------------------------------------------------
// WMMA compute for one 32-wide K slice from LDS.
// A fragment (ISA 16-bit A 16x32): lanes 0-15 hold K{0..7,16..23}, lanes
// 16-31 hold K{8..15,24..31}, row = lane&15.  B fragment (32x16): lanes 0-15
// col=lane, K{0..7,16..23}; lanes 16-31 K{8..15,24..31}.
// ---------------------------------------------------------------------------
__device__ __forceinline__ void wmma_tile(const _Float16 (*lA)[40],
                                          const _Float16 (*lB)[40],
                                          int wm, int wn, int half, int l16,
                                          v8f (*acc)[2]) {
    v16h afrag[4], bfrag[2];
    #pragma unroll
    for (int mi = 0; mi < 4; ++mi) {
        const _Float16* pa = &lA[wm + mi * 16 + l16][half * 8];
        frag16 f;
        f.q[0] = *(const uint4*)pa;         // K base .. base+7
        f.q[1] = *(const uint4*)(pa + 16);  // K base+16 .. base+23
        afrag[mi] = f.v;
    }
    #pragma unroll
    for (int ni = 0; ni < 2; ++ni) {
        const _Float16* pb = &lB[wn + ni * 16 + l16][half * 8];
        frag16 f;
        f.q[0] = *(const uint4*)pb;
        f.q[1] = *(const uint4*)(pb + 16);
        bfrag[ni] = f.v;
    }
    #pragma unroll
    for (int mi = 0; mi < 4; ++mi)
        #pragma unroll
        for (int ni = 0; ni < 2; ++ni)
            acc[mi][ni] = __builtin_amdgcn_wmma_f32_16x16x32_f16(
                false, afrag[mi], false, bfrag[ni],
                (short)0, acc[mi][ni], false, false);
}

// ---------------------------------------------------------------------------
// Pass 2: out[b](512x4096) = W(512x512) * S[b](512x4096) + bias, f16 WMMA.
// Block tile 128(M) x 128(N); 8 waves as 2(M) x 4(N); wave tile 64x32 =
// 4x2 subtiles of v_wmma_f32_16x16x32_f16.  K loop: 16 steps of 32.
// Async path: double-buffered LDS fed by global_load_async_to_lds_b128,
// synchronized with s_wait_asynccnt + workgroup barrier.
// ---------------------------------------------------------------------------
__global__ __launch_bounds__(256) void gemm_kernel(const _Float16* __restrict__ Wh,
                                                   const _Float16* __restrict__ St,
                                                   const float*    __restrict__ bias,
                                                   float*          __restrict__ out) {
    __shared__ _Float16 ldsA[2][128][40];   // [buf][m][k], rows padded to 40
    __shared__ _Float16 ldsB[2][128][40];   // [buf][n][k]

    const int b    = blockIdx.z;
    const int n0   = blockIdx.x * 128;
    const int m0   = blockIdx.y * 128;
    const int t    = threadIdx.x;
    const int lane = t & 31;
    const int wv   = t >> 5;
    const int wm   = (wv >> 2) * 64;     // wave M offset within block tile
    const int wn   = (wv & 3) * 32;      // wave N offset within block tile

    // Staging: each thread moves 32B per matrix; thread pair = one 64B row.
    const int row  = t >> 1;
    const int colh = (t & 1) * 16;       // f16 offset within 32-wide K tile

    const _Float16* gA = Wh + (size_t)(m0 + row) * kC + colh;
    const _Float16* gB = St + ((size_t)b * kN + (size_t)(n0 + row)) * kC + colh;

    v8f acc[4][2];
    #pragma unroll
    for (int mi = 0; mi < 4; ++mi)
        #pragma unroll
        for (int ni = 0; ni < 2; ++ni)
            acc[mi][ni] = (v8f)0.0f;

    const int half = lane >> 4;          // 0: lanes 0-15, 1: lanes 16-31
    const int l16  = lane & 15;

#if USE_ASYNC_COPY
    // imm offset 16 applies to BOTH global and LDS addresses (ISA §10.7
    // async pseudocode), covering the second b128 of each 32B slice.
    auto issue = [&](int buf, int k0) {
        __builtin_amdgcn_global_load_async_to_lds_b128(
            to_global(gA + k0), to_lds(&ldsA[buf][row][colh]), 0, 0);
        __builtin_amdgcn_global_load_async_to_lds_b128(
            to_global(gA + k0), to_lds(&ldsA[buf][row][colh]), 16, 0);
        __builtin_amdgcn_global_load_async_to_lds_b128(
            to_global(gB + k0), to_lds(&ldsB[buf][row][colh]), 0, 0);
        __builtin_amdgcn_global_load_async_to_lds_b128(
            to_global(gB + k0), to_lds(&ldsB[buf][row][colh]), 16, 0);
    };
    issue(0, 0);
    for (int kt = 0; kt < kC / 32; ++kt) {
        const int cur = kt & 1;
        if (kt + 1 < kC / 32) {
            issue(cur ^ 1, (kt + 1) * 32);   // prefetch next tile, other buffer
            // <=4 outstanding: the batch just issued; current tile is done.
#if __has_builtin(__builtin_amdgcn_s_wait_asynccnt)
            __builtin_amdgcn_s_wait_asynccnt(4);
#else
            asm volatile("s_wait_asynccnt 0x4" ::: "memory");
#endif
        } else {
#if __has_builtin(__builtin_amdgcn_s_wait_asynccnt)
            __builtin_amdgcn_s_wait_asynccnt(0);
#else
            asm volatile("s_wait_asynccnt 0x0" ::: "memory");
#endif
        }
        __syncthreads();                     // all waves' slices landed
        wmma_tile(ldsA[cur], ldsB[cur], wm, wn, half, l16, acc);
        __syncthreads();                     // done reading before overwrite
    }
#else
    // Fallback: register-prefetch pipeline into single LDS buffer.
    uint4 ra0 = *(const uint4*)(gA);
    uint4 ra1 = *(const uint4*)(gA + 8);
    uint4 rb0 = *(const uint4*)(gB);
    uint4 rb1 = *(const uint4*)(gB + 8);
    for (int k0 = 0; k0 < kC; k0 += 32) {
        *(uint4*)&ldsA[0][row][colh]     = ra0;
        *(uint4*)&ldsA[0][row][colh + 8] = ra1;
        *(uint4*)&ldsB[0][row][colh]     = rb0;
        *(uint4*)&ldsB[0][row][colh + 8] = rb1;
        __syncthreads();
        if (k0 + 32 < kC) {
            ra0 = *(const uint4*)(gA + k0 + 32);
            ra1 = *(const uint4*)(gA + k0 + 40);
            rb0 = *(const uint4*)(gB + k0 + 32);
            rb1 = *(const uint4*)(gB + k0 + 40);
        }
        wmma_tile(ldsA[0], ldsB[0], wm, wn, half, l16, acc);
        __syncthreads();
    }
#endif

    // Epilogue: C/D layout — lane L: col = L&15, VGPR r -> row r + (L>>4)*8.
    #pragma unroll
    for (int mi = 0; mi < 4; ++mi) {
        const int mrow = m0 + wm + mi * 16 + half * 8;
        #pragma unroll
        for (int ni = 0; ni < 2; ++ni) {
            const int ncol = n0 + wn + ni * 16 + l16;
            float* po = out + ((size_t)b * kOUT + (size_t)mrow) * kN + ncol;
            #pragma unroll
            for (int r = 0; r < 8; ++r)
                po[(size_t)r * kN] = acc[mi][ni][r] + bias[mrow + r];
        }
    }
}

// ---------------------------------------------------------------------------
// Launch: inputs are (x, offset, weight, bias, attn_mask); output f32 [B,OUT,N].
// ws layout: St f16 [B][N][C] (32 MiB) then Wh f16 [OUT][C] (0.5 MiB).
// ---------------------------------------------------------------------------
extern "C" void kernel_launch(void* const* d_in, const int* in_sizes, int n_in,
                              void* d_out, int out_size, void* d_ws, size_t ws_size,
                              hipStream_t stream) {
    const float* x      = (const float*)d_in[0];
    const float* offset = (const float*)d_in[1];
    const float* weight = (const float*)d_in[2];
    const float* bias   = (const float*)d_in[3];
    const int*   mask   = (const int*)d_in[4];
    float* out = (float*)d_out;

    _Float16* St = (_Float16*)d_ws;                        // kB*kN*kC halves
    _Float16* Wh = St + (size_t)kB * kN * kC;              // kOUT*kC halves

    wcvt_kernel<<<(kOUT * kC) / 256, 256, 0, stream>>>(weight, Wh);
    sample_kernel<<<dim3(kN / 64, kC / 64, kB), 256, 0, stream>>>(x, offset, mask, St);
    gemm_kernel<<<dim3(kN / 128, kOUT / 128, kB), 256, 0, stream>>>(Wh, St, bias, out);
}
